// detection_head_52802327937436
// MI455X (gfx1250) — matrix operand order repaired and checked by reading, not compiled, not run on hardware
//
#include <hip/hip_runtime.h>

// CDNA5 / gfx1250 WMMA types
typedef __attribute__((ext_vector_type(16))) __bf16 v16bf;
typedef __attribute__((ext_vector_type(8)))  float  v8f;

namespace {
constexpr int kHW   = 252;
constexpr int kB    = 4;
constexpr int kCin  = 384;
constexpr int kPix  = kB * kHW * kHW;   // 254016 pixels
constexpr int kNout = 52;               // 4+12+4+12+4+16 channels
constexpr int kKsteps  = kCin / 32;     // 12 K-steps of 32 (bf16 WMMA K)
constexpr int kMTiles  = 2;             // 16-row M tiles per wave
constexpr int kMBlock  = 8 * 16 * kMTiles;  // 256 rows per block (8 waves)
constexpr int kThreads = 256;

// Pack two f32 into packed bf16 (truncation) with a single v_perm_b32:
// result = { hi.bits[31:16], lo.bits[31:16] }
__device__ __forceinline__ unsigned pack2_bf16(float lo, float hi) {
  return __builtin_amdgcn_perm(__float_as_uint(hi), __float_as_uint(lo),
                               0x07060302u);
}
// one-time weight conversion: round-to-nearest-even
__device__ __forceinline__ unsigned short cvt_bf16(float a) {
  unsigned ua = __float_as_uint(a);
  return (unsigned short)((ua + 0x7FFFu + ((ua >> 16) & 1u)) >> 16);
}

// Epilogue: bias + sigmoid(occ/heading) + scatter into concatenated outputs.
// D layout: lane holds column N = lane&15; VGPR r holds row M = r + (lane>=16 ? 8:0)
template <bool TAIL>
__device__ __forceinline__ void store_rows(const v8f* accT, int mBase, int col,
                                           const float* bias64,
                                           float* __restrict__ out) {
#pragma unroll
  for (int n = 0; n < 4; ++n) {
    const int o = n * 16 + col;
    if (o >= kNout) continue;  // padded channels 52..63 (n==3, col>=4)
    size_t base; int cc, oc;
    if      (o < 4)  { base = (size_t)0;          cc = 4;  oc = o;      }
    else if (o < 16) { base = (size_t)4  * kPix;  cc = 12; oc = o - 4;  }
    else if (o < 20) { base = (size_t)16 * kPix;  cc = 4;  oc = o - 16; }
    else if (o < 32) { base = (size_t)20 * kPix;  cc = 12; oc = o - 20; }
    else if (o < 36) { base = (size_t)32 * kPix;  cc = 4;  oc = o - 32; }
    else             { base = (size_t)36 * kPix;  cc = 16; oc = o - 36; }
    const float bias = bias64[o];
    // sigmoid channels: occ = 0..3 (tile 0, col<4), heading = 32..35 (tile 2, col<4)
    const bool sig = (n == 0 || n == 2) && (col < 4);
#pragma unroll
    for (int r = 0; r < 8; ++r) {
      const int pix = mBase + r;
      float v = accT[n][r] + bias;
      if (n == 0 || n == 2) {  // compile-time pruned for n==1,3
        const float s = __builtin_amdgcn_rcpf(1.0f + __expf(-v));
        v = sig ? s : v;       // branchless select, no exec juggling
      }
      if (!TAIL || pix < kPix)
        out[base + (size_t)pix * cc + oc] = v;
    }
  }
}
} // namespace

__global__ __launch_bounds__(kThreads)
void det_head_wmma(const float* __restrict__ x,
                   const float* __restrict__ occ_w,  const float* __restrict__ occ_b,
                   const float* __restrict__ loc_w,  const float* __restrict__ loc_b,
                   const float* __restrict__ angle_w,const float* __restrict__ angle_b,
                   const float* __restrict__ size_w, const float* __restrict__ size_b,
                   const float* __restrict__ head_w, const float* __restrict__ head_b,
                   const float* __restrict__ clf_w,  const float* __restrict__ clf_b,
                   float* __restrict__ out)
{
  // Weights pre-packed as bf16 B-fragments: [ks][ntile][lane][16 halfwords] = 48 KB
  __shared__ __align__(16) unsigned short lds_w[kKsteps * 4 * 32 * 16];
  __shared__ float lds_bias[64];

  const int tid  = threadIdx.x;
  const int lane = tid & 31;
  const int wave = tid >> 5;

  // ---- stage weights into LDS in ISA B-matrix (32x16 bf16) fragment layout ----
  // B layout: lane l holds column n = l&15; halfword h maps to K = ((h&8)<<1) + ((l>>4)<<3) + (h&7)
  for (int idx = tid; idx < kKsteps * 4 * 32 * 16; idx += kThreads) {
    const int h  = idx & 15;
    const int l  = (idx >> 4) & 31;
    const int n  = (idx >> 9) & 3;
    const int ks = idx >> 11;
    const int o  = n * 16 + (l & 15);
    const int k  = ks * 32 + ((h & 8) << 1) + ((l >> 4) << 3) + (h & 7);
    float v = 0.f;
    if      (o < 4)  v = occ_w[o * kCin + k];
    else if (o < 16) v = loc_w[(o - 4)  * kCin + k];
    else if (o < 20) v = angle_w[(o - 16) * kCin + k];
    else if (o < 32) v = size_w[(o - 20) * kCin + k];
    else if (o < 36) v = head_w[(o - 32) * kCin + k];
    else if (o < 52) v = clf_w[(o - 36) * kCin + k];
    lds_w[idx] = cvt_bf16(v);
  }
  if (tid < 64) {
    const int o = tid;
    float v = 0.f;
    if      (o < 4)  v = occ_b[o];
    else if (o < 16) v = loc_b[o - 4];
    else if (o < 20) v = angle_b[o - 16];
    else if (o < 32) v = size_b[o - 20];
    else if (o < 36) v = head_b[o - 32];
    else if (o < 52) v = clf_b[o - 36];
    lds_bias[o] = v;
  }
  __syncthreads();

  // ---- per-wave: kMTiles x 16(M=pixels) by 64(N=channels) ----
  const int waveRow = blockIdx.x * kMBlock + wave * (16 * kMTiles);
  const int klo = (lane >> 4) << 3;  // A layout: lanes>=16 start at K+8
  const float* __restrict__ xrow[kMTiles];
#pragma unroll
  for (int t = 0; t < kMTiles; ++t) {
    int r = waveRow + t * 16 + (lane & 15);
    if (r > kPix - 1) r = kPix - 1;  // clamp loads in ragged tail
    xrow[t] = x + (size_t)r * kCin + klo;
  }

  v8f acc[kMTiles][4] = {};

  for (int ks = 0; ks < kKsteps; ++ks) {
    // A fragments (16x32 bf16): lane<16 holds K {0..7,16..23}, lane>=16 {8..15,24..31}
    union { v16bf v; unsigned u[8]; } a[kMTiles];
#pragma unroll
    for (int t = 0; t < kMTiles; ++t) {
      const float4* xp = (const float4*)(xrow[t] + ks * 32);
      const float4 f0 = xp[0];
      const float4 f1 = xp[1];
      const float4 f2 = xp[4];
      const float4 f3 = xp[5];
      a[t].u[0] = pack2_bf16(f0.x, f0.y); a[t].u[1] = pack2_bf16(f0.z, f0.w);
      a[t].u[2] = pack2_bf16(f1.x, f1.y); a[t].u[3] = pack2_bf16(f1.z, f1.w);
      a[t].u[4] = pack2_bf16(f2.x, f2.y); a[t].u[5] = pack2_bf16(f2.z, f2.w);
      a[t].u[6] = pack2_bf16(f3.x, f3.y); a[t].u[7] = pack2_bf16(f3.z, f3.w);
    }
#pragma unroll
    for (int n = 0; n < 4; ++n) {
      union { v16bf v; uint4 q[2]; } b;
      const uint4* bp = (const uint4*)&lds_w[((ks * 4 + n) * 32 + lane) * 16];
      b.q[0] = bp[0];
      b.q[1] = bp[1];
#pragma unroll
      for (int t = 0; t < kMTiles; ++t) {
        // (neg_a, A, neg_b, B, c_mod, C, reuse_a, reuse_b)
        acc[t][n] = __builtin_amdgcn_wmma_f32_16x16x32_bf16(
            false, a[t].v, false, b.v, (short)0, acc[t][n], false, false);
      }
    }
  }

  const int col  = lane & 15;
  const bool tail = (waveRow + 16 * kMTiles) > kPix;
#pragma unroll
  for (int t = 0; t < kMTiles; ++t) {
    const int mBase = waveRow + t * 16 + ((lane >> 4) << 3);
    if (tail) store_rows<true >(acc[t], mBase, col, lds_bias, out);
    else      store_rows<false>(acc[t], mBase, col, lds_bias, out);
  }
}

extern "C" void kernel_launch(void* const* d_in, const int* in_sizes, int n_in,
                              void* d_out, int out_size, void* d_ws, size_t ws_size,
                              hipStream_t stream) {
  const float* x       = (const float*)d_in[0];
  const float* occ_w   = (const float*)d_in[1];
  const float* occ_b   = (const float*)d_in[2];
  const float* loc_w   = (const float*)d_in[3];
  const float* loc_b   = (const float*)d_in[4];
  const float* angle_w = (const float*)d_in[5];
  const float* angle_b = (const float*)d_in[6];
  const float* size_w  = (const float*)d_in[7];
  const float* size_b  = (const float*)d_in[8];
  const float* head_w  = (const float*)d_in[9];
  const float* head_b  = (const float*)d_in[10];
  const float* clf_w   = (const float*)d_in[11];
  const float* clf_b   = (const float*)d_in[12];

  const int grid = (kPix + kMBlock - 1) / kMBlock;   // 993 blocks of 8 waves
  det_head_wmma<<<grid, kThreads, 0, stream>>>(
      x, occ_w, occ_b, loc_w, loc_b, angle_w, angle_b,
      size_w, size_b, head_w, head_b, clf_w, clf_b, (float*)d_out);
}
